// DGCNN_sortpool_mean_7842610283368
// MI455X (gfx1250) — matrix operand order repaired and checked by reading, not compiled, not run on hardware
//
#include <hip/hip_runtime.h>
#include <hip/hip_bf16.h>
#include <math.h>

// Problem constants (from reference)
#define NN 131072   // nodes
#define EE 1048576  // edges
#define SS 4096     // subgraphs
#define GG 128      // graphs
#define FF 128      // input feat
#define HH 256      // hidden
#define KK 30       // sortpool k
#define DDIM 1025   // H*L+1 concat dim
#define CDIV(a,b) (((a)+(b)-1)/(b))

typedef __attribute__((ext_vector_type(16))) _Float16 v16h;
typedef __attribute__((ext_vector_type(8)))  _Float16 v8h;
typedef __attribute__((ext_vector_type(8)))  float    v8f;

// ---------------- small utility kernels ----------------
__global__ void k_zero(float* p, int n){ int t=blockIdx.x*blockDim.x+threadIdx.x; if(t<n) p[t]=0.f; }
__global__ void k_fill1(float* p, int n){ int t=blockIdx.x*blockDim.x+threadIdx.x; if(t<n) p[t]=1.f; }

__global__ void k_deg(const int* __restrict__ dst, float* __restrict__ dinv){
  int e=blockIdx.x*blockDim.x+threadIdx.x; if(e<EE) atomicAdd(&dinv[dst[e]], 1.f);
}
__global__ void k_rsqrt(float* dinv){ int n=blockIdx.x*blockDim.x+threadIdx.x; if(n<NN) dinv[n]=rsqrtf(dinv[n]); }
__global__ void k_coef(const int* __restrict__ src, const int* __restrict__ dst,
                       const float* __restrict__ dinv, float* __restrict__ coef){
  int e=blockIdx.x*blockDim.x+threadIdx.x; if(e<EE) coef[e]=dinv[src[e]]*dinv[dst[e]];
}
__global__ void k_f32tof16(const float* __restrict__ x, _Float16* __restrict__ y, long n){
  long t=blockIdx.x*(long)blockDim.x+threadIdx.x; if(t<n) y[t]=(_Float16)x[t];
}
// convert W (Kin x Nout, row major f32) -> Wt (Nout x Kin, f16) so B fragments load contiguously
__global__ void k_wtrans(const float* __restrict__ W, _Float16* __restrict__ Wt, int Kin, int Nout){
  int t=blockIdx.x*blockDim.x+threadIdx.x; if(t>=Kin*Nout) return;
  int k=t/Nout, o=t%Nout; Wt[(size_t)o*Kin+k]=(_Float16)W[t];
}

// ---------------- WMMA GEMM: C(MxNout,f32) = A(MxKin,f16) * Bt^T ----------------
// One wave computes a 16x32 tile: shared A fragment, two 16x16 WMMAs per K-step.
__global__ __launch_bounds__(256) void k_gemm(const _Float16* __restrict__ A,
                                              const _Float16* __restrict__ Bt,
                                              float* __restrict__ C,
                                              int M, int Kin, int Nout){
  int lane = threadIdx.x & 31;
  int wid  = blockIdx.x*(blockDim.x>>5) + (threadIdx.x>>5);
  int ntile2 = Nout >> 5;                       // 16x32 tiles per row
  int tm = wid / ntile2;
  int tn = (wid % ntile2) << 5;
  if (tm*16 >= M) return;

  int row  = tm*16 + (lane & 15);
  int aoff = (lane < 16) ? 0 : 16;              // contiguous k order, matches B
  int koff = (lane < 16) ? 0 : 16;              // contiguous k order, matches A
  const _Float16* arow = A  + (size_t)row*Kin;
  const _Float16* b0p  = Bt + (size_t)(tn      + (lane&15))*Kin + koff;
  const _Float16* b1p  = Bt + (size_t)(tn + 16 + (lane&15))*Kin + koff;

  v8f c0 = {}; v8f c1 = {};
  for (int k0 = 0; k0 < Kin; k0 += 32){
    v8h lo = *(const v8h*)(arow + k0 + aoff);
    v8h hi = *(const v8h*)(arow + k0 + 8 + aoff);
    v16h a;
    #pragma unroll
    for (int i=0;i<8;++i){ a[i]=lo[i]; a[i+8]=hi[i]; }
    v16h b0 = *(const v16h*)(b0p + k0);
    v16h b1 = *(const v16h*)(b1p + k0);
    c0 = __builtin_amdgcn_wmma_f32_16x16x32_f16(false, a, false, b0, (short)0, c0, false, false);
    c1 = __builtin_amdgcn_wmma_f32_16x16x32_f16(false, a, false, b1, (short)0, c1, false, false);
  }
  int mb  = tm*16 + ((lane>=16)?8:0);           // ISA f32 C/D layout
  int col = lane & 15;
  #pragma unroll
  for (int r=0;r<8;++r){
    C[(size_t)(mb+r)*Nout + tn + col]      = c0[r];
    C[(size_t)(mb+r)*Nout + tn + 16 + col] = c1[r];
  }
}

// Final GCN layer projection (HH -> 1): one wave per node
__global__ void k_gemv1(const _Float16* __restrict__ xh, const float* __restrict__ W4,
                        float* __restrict__ xw1){
  int wid = blockIdx.x*(blockDim.x>>5) + (threadIdx.x>>5);
  int lane = threadIdx.x & 31;
  if (wid >= NN) return;
  const _Float16* row = xh + (size_t)wid*HH;
  float a = 0.f;
  for (int k=lane; k<HH; k+=32) a += (float)row[k]*W4[k];
  for (int off=16; off>0; off>>=1) a += __shfl_xor(a, off, 32);
  if (lane==0) xw1[wid]=a;
}

// ---------------- message passing ----------------
template<int DIM>
__global__ void k_selfinit(const float* __restrict__ xw, const float* __restrict__ dinv,
                           float* __restrict__ agg, long total){
  long t = blockIdx.x*(long)blockDim.x + threadIdx.x;
  if (t>=total) return;
  int n = (int)(t / DIM);
  float d = dinv[n];
  agg[t] = xw[t]*d*d;
}
template<int DIM>
__global__ void k_scatter(const int* __restrict__ src, const int* __restrict__ dst,
                          const float* __restrict__ coef, const float* __restrict__ xw,
                          float* __restrict__ agg){
  long t = blockIdx.x*(long)blockDim.x + threadIdx.x;
  if (t >= (long)EE*DIM) return;
  int e = (int)(t / DIM);
  int f = (int)(t % DIM);
  atomicAdd(&agg[(size_t)dst[e]*DIM + f], xw[(size_t)src[e]*DIM + f]*coef[e]);
}
template<int DIM, bool WX>
__global__ void k_act(const float* __restrict__ agg, const float* __restrict__ b,
                      float* __restrict__ cc, _Float16* __restrict__ xh, int colOff){
  long t = blockIdx.x*(long)blockDim.x + threadIdx.x;
  if (t >= (long)NN*DIM) return;
  int n = (int)(t/DIM), f = (int)(t%DIM);
  float v = tanhf(agg[t] + b[f]);
  cc[(size_t)n*DDIM + colOff + f] = v;
  if (WX) xh[t] = (_Float16)v;
}

// ---------------- sort-pool ----------------
// node_to_subgraph is sorted -> each subgraph is a contiguous node range; binary search starts
__global__ void k_starts(const int* __restrict__ sub, int* __restrict__ starts){
  int s = blockIdx.x*blockDim.x + threadIdx.x;
  if (s > SS) return;
  int lo=0, hi=NN;
  while (lo<hi){ int mid=(lo+hi)>>1; if (sub[mid] < s) lo=mid+1; else hi=mid; }
  starts[s]=lo;
}
// top-K by (last feature desc, node index asc) — matches stable lexsort
__global__ void k_topk(const float* __restrict__ cc, const int* __restrict__ starts,
                       int* __restrict__ topk){
  int s = blockIdx.x*blockDim.x + threadIdx.x;
  if (s >= SS) return;
  int beg = starts[s], end = starts[s+1];
  float pv = INFINITY; int pi = -1;
  for (int r=0;r<KK;++r){
    float bv = -INFINITY; int bi = -1;
    for (int n=beg;n<end;++n){
      float v = cc[(size_t)n*DDIM + (DDIM-1)];
      bool after  = (v < pv) || (v==pv && n>pi);
      if (!after) continue;
      bool better = (v > bv) || (v==bv && n<bi);
      if (better){ bv=v; bi=n; }
    }
    topk[s*KK+r]=bi;
    if (bi<0){ for(int r2=r+1;r2<KK;++r2) topk[s*KK+r2]=-1; return; }
    pv=bv; pi=bi;
  }
}

// conv1: stride-D conv == per-rank dense (D=1025 -> 16). One wave per (s, rank); w1 in LDS (f16)
__global__ __launch_bounds__(256) void k_conv1(const float* __restrict__ cc, const int* __restrict__ topk,
                        const float* __restrict__ w1, const float* __restrict__ b1,
                        float* __restrict__ y1){
  __shared__ _Float16 sw[16*DDIM];          // 32.8 KB
  for (int i=threadIdx.x; i<16*DDIM; i+=blockDim.x) sw[i]=(_Float16)w1[i];
  __syncthreads();
  int wid = blockIdx.x*(blockDim.x>>5) + (threadIdx.x>>5);
  int lane = threadIdx.x & 31;
  if (wid >= SS*KK) return;
  int s = wid / KK, kpos = wid % KK;
  int node = topk[s*KK+kpos];
  float acc[16];
  #pragma unroll
  for (int o=0;o<16;++o) acc[o]=0.f;
  if (node >= 0){
    const float* row = cc + (size_t)node*DDIM;
    for (int d=lane; d<DDIM; d+=32){
      float v = row[d];
      #pragma unroll
      for (int o=0;o<16;++o) acc[o] += v * (float)sw[o*DDIM+d];
    }
  }
  #pragma unroll
  for (int o=0;o<16;++o){
    float a = acc[o];
    for (int off=16; off>0; off>>=1) a += __shfl_xor(a, off, 32);
    if (lane==0) y1[(size_t)s*(16*KK) + o*KK + kpos] = fmaxf(a + b1[o], 0.f);
  }
}

__global__ void k_maxpool(const float* __restrict__ y1, float* __restrict__ y2){
  int t = blockIdx.x*blockDim.x + threadIdx.x;
  if (t >= SS*16*15) return;
  int s=t/240, r=t%240, o=r/15, j=r%15;
  float a=y1[(size_t)s*480 + o*30 + 2*j], b=y1[(size_t)s*480 + o*30 + 2*j+1];
  y2[t]=fmaxf(a,b);
}
__global__ void k_conv2(const float* __restrict__ y2, const float* __restrict__ w2,
                        const float* __restrict__ b2, float* __restrict__ y3){
  int t = blockIdx.x*blockDim.x + threadIdx.x;
  if (t >= SS*352) return;
  int s=t/352, r=t%352, o2=r/11, tt=r%11;
  float a=b2[o2];
  #pragma unroll
  for (int i=0;i<16;++i)
    #pragma unroll
    for (int q=0;q<5;++q)
      a += y2[(size_t)s*240 + i*15 + tt + q] * w2[o2*80 + i*5 + q];
  y3[t]=fmaxf(a,0.f);
}
__global__ void k_segsum(const float* __restrict__ y3, const int* __restrict__ s2g,
                         float* __restrict__ gsum){
  int t = blockIdx.x*blockDim.x + threadIdx.x;
  if (t >= SS*352) return;
  int s=t/352, f=t%352;
  atomicAdd(&gsum[s2g[s]*352+f], y3[t]);
}
__global__ void k_segcnt(const int* __restrict__ s2g, float* __restrict__ gcnt){
  int s = blockIdx.x*blockDim.x + threadIdx.x;
  if (s<SS) atomicAdd(&gcnt[s2g[s]], 1.f);
}
__global__ void k_lin1(const float* __restrict__ gsum, const float* __restrict__ gcnt,
                       const float* __restrict__ w, const float* __restrict__ b,
                       float* __restrict__ hid){
  int t = blockIdx.x*blockDim.x + threadIdx.x;
  if (t >= GG*128) return;
  int g=t/128, j=t%128;
  float inv = 1.f / fmaxf(gcnt[g], 1.f);
  float a = b[j];
  for (int f=0; f<352; ++f) a += gsum[g*352+f]*inv*w[f*128+j];
  hid[t]=fmaxf(a,0.f);
}
__global__ void k_lin2(const float* __restrict__ hid, const float* __restrict__ w,
                       const float* __restrict__ b, float* __restrict__ out){
  int g = blockIdx.x*blockDim.x + threadIdx.x;
  if (g >= GG) return;
  float l0=b[0], l1=b[1];
  for (int j=0;j<128;++j){ float h=hid[g*128+j]; l0+=h*w[j*2]; l1+=h*w[j*2+1]; }
  float m=fmaxf(l0,l1);
  float lse = m + logf(expf(l0-m)+expf(l1-m));
  out[g*2]=l0-lse; out[g*2+1]=l1-lse;
}

extern "C" void kernel_launch(void* const* d_in, const int* in_sizes, int n_in,
                              void* d_out, int out_size, void* d_ws, size_t ws_size,
                              hipStream_t stream) {
  (void)in_sizes; (void)n_in; (void)out_size; (void)ws_size;
  const float* x   = (const float*)d_in[0];
  const int*   ei  = (const int*)d_in[1];
  const int*   src = ei;
  const int*   dst = ei + EE;
  const int*   n2s = (const int*)d_in[2];
  const int*   s2g = (const int*)d_in[3];
  const float* Wc[5]; for (int i=0;i<5;++i) Wc[i]=(const float*)d_in[4+i];
  const float* bc[5]; for (int i=0;i<5;++i) bc[i]=(const float*)d_in[9+i];
  const float* w1=(const float*)d_in[14]; const float* b1=(const float*)d_in[15];
  const float* w2=(const float*)d_in[16]; const float* b2=(const float*)d_in[17];
  const float* l1w=(const float*)d_in[18]; const float* l1b=(const float*)d_in[19];
  const float* l2w=(const float*)d_in[20]; const float* l2b=(const float*)d_in[21];
  float* out = (float*)d_out;

  // workspace carve-out (256B aligned)
  char* ws = (char*)d_ws; size_t off = 0;
  auto alloc = [&](size_t bytes)->char*{ char* p = ws + off; off += (bytes + 255) & ~(size_t)255; return p; };
  float*    dinv   = (float*)alloc((size_t)NN*4);
  float*    coef   = (float*)alloc((size_t)EE*4);
  _Float16* xh     = (_Float16*)alloc((size_t)NN*HH*2);
  float*    xw     = (float*)alloc((size_t)NN*HH*4);
  float*    agg    = (float*)alloc((size_t)NN*HH*4);
  float*    cc     = (float*)alloc((size_t)NN*DDIM*4);
  _Float16* Wt     = (_Float16*)alloc((size_t)HH*HH*2);
  int*      starts = (int*)alloc((size_t)(SS+1)*4);
  int*      topk   = (int*)alloc((size_t)SS*KK*4);
  float*    y1     = (float*)alloc((size_t)SS*16*KK*4);
  float*    y2     = (float*)alloc((size_t)SS*16*15*4);
  float*    y3     = (float*)alloc((size_t)SS*352*4);
  float*    gsum   = (float*)alloc((size_t)GG*352*4);
  float*    gcnt   = (float*)alloc((size_t)GG*4);
  float*    hid    = (float*)alloc((size_t)GG*128*4);

  const int B = 256;
  // zero accumulators (re-init every call; deterministic)
  k_zero<<<CDIV(GG*352,B),B,0,stream>>>(gsum, GG*352);
  k_zero<<<CDIV(GG,B),B,0,stream>>>(gcnt, GG);

  // degree / normalization (shared by all 5 GCN layers)
  k_fill1<<<CDIV(NN,B),B,0,stream>>>(dinv, NN);
  k_deg<<<CDIV(EE,B),B,0,stream>>>(dst, dinv);
  k_rsqrt<<<CDIV(NN,B),B,0,stream>>>(dinv);
  k_coef<<<CDIV(EE,B),B,0,stream>>>(src, dst, dinv, coef);

  // f16 copy of input features for WMMA A operand
  k_f32tof16<<<(int)CDIV((long)NN*FF,B),B,0,stream>>>(x, xh, (long)NN*FF);

  // GCN layers 0..3 (WMMA GEMM + scatter + tanh)
  for (int l=0; l<4; ++l){
    int Kin = (l==0)? FF : HH;
    k_wtrans<<<CDIV(Kin*HH,B),B,0,stream>>>(Wc[l], Wt, Kin, HH);
    int waves = (NN/16)*(HH/32);
    k_gemm<<<waves/8,B,0,stream>>>(xh, Wt, xw, NN, Kin, HH);
    long tot = (long)NN*HH;
    k_selfinit<HH><<<(int)CDIV(tot,B),B,0,stream>>>(xw, dinv, agg, tot);
    k_scatter<HH><<<(int)CDIV((long)EE*HH,B),B,0,stream>>>(src, dst, coef, xw, agg);
    k_act<HH,true><<<(int)CDIV(tot,B),B,0,stream>>>(agg, bc[l], cc, xh, l*HH);
  }
  // GCN layer 4 (HH -> 1)
  k_gemv1<<<NN/8,B,0,stream>>>(xh, Wc[4], xw);
  k_selfinit<1><<<CDIV(NN,B),B,0,stream>>>(xw, dinv, agg, NN);
  k_scatter<1><<<CDIV(EE,B),B,0,stream>>>(src, dst, coef, xw, agg);
  k_act<1,false><<<CDIV(NN,B),B,0,stream>>>(agg, bc[4], cc, nullptr, 4*HH);

  // sort-pool
  k_starts<<<CDIV(SS+1,B),B,0,stream>>>(n2s, starts);
  k_topk<<<CDIV(SS,B),B,0,stream>>>(cc, starts, topk);

  // readout head
  k_conv1<<<SS*KK/8,B,0,stream>>>(cc, topk, w1, b1, y1);
  k_maxpool<<<CDIV(SS*16*15,B),B,0,stream>>>(y1, y2);
  k_conv2<<<CDIV(SS*352,B),B,0,stream>>>(y2, w2, b2, y3);
  k_segsum<<<CDIV(SS*352,B),B,0,stream>>>(y3, s2g, gsum);
  k_segcnt<<<CDIV(SS,B),B,0,stream>>>(s2g, gcnt);
  k_lin1<<<CDIV(GG*128,B),B,0,stream>>>(gsum, gcnt, l1w, l1b, hid);
  k_lin2<<<CDIV(GG,B),B,0,stream>>>(hid, l2w, l2b, out);
}
